// Int8Linear_37580963840179
// MI455X (gfx1250) — matrix operand order repaired and checked by reading, not compiled, not run on hardware
//
#include <hip/hip_runtime.h>
#include <stdint.h>

#ifndef __has_builtin
#define __has_builtin(x) 0
#endif

// ---------------------------------------------------------------------------
// int8-weight linear:  out = scale * (x @ qweight^T) + bias
//   x: [M=16384, K=4096] fp32, qweight: [N=4096, K=4096] int32 (int8 values)
// Strategy: int8 is exact in fp16 -> one-pass conversions into d_ws, then a
// double-buffered fp16 WMMA GEMM (v_wmma_f32_16x16x32_f16, fp32 accum) with
// async global->LDS copies (ASYNCcnt) when the toolchain exposes them.
// ---------------------------------------------------------------------------

typedef _Float16 half_t;
typedef __attribute__((ext_vector_type(16))) _Float16 v16h;
typedef __attribute__((ext_vector_type(8)))  _Float16 v8h;
typedef __attribute__((ext_vector_type(8)))  float    v8f;

// The async-LDS builtin's pointee type is a GCC-style vector:
//   __attribute__((__vector_size__(16))) int   (AS1 source, AS3 dest)
typedef int vsi4 __attribute__((vector_size(16)));
typedef __attribute__((address_space(1))) vsi4* g4p;
typedef __attribute__((address_space(3))) vsi4* l4p;

constexpr int M_DIM = 8 * 2048;   // 16384
constexpr int N_DIM = 4096;
constexpr int K_DIM = 4096;

constexpr int BM = 256;            // block tile M
constexpr int BN = 128;            // block tile N
constexpr int BK = 32;             // K-step (one WMMA K depth)
constexpr int LDSTR = BK + 8;      // padded LDS stride (40 halves = 80 B)

#if __has_builtin(__builtin_amdgcn_global_load_async_to_lds_b128)
#define USE_ASYNC_LDS 1
#else
#define USE_ASYNC_LDS 0
#endif

__device__ __forceinline__ void wait_asynccnt0() {
#if __has_builtin(__builtin_amdgcn_s_wait_asynccnt)
    __builtin_amdgcn_s_wait_asynccnt(0);
#else
    asm volatile("s_wait_asynccnt 0x0" ::: "memory");
#endif
}

__device__ __forceinline__ g4p as_g4(const void* p) {
    return (g4p)(uintptr_t)p;                     // AS1, const dropped via int
}
__device__ __forceinline__ l4p as_l4(void* p) {
    return (l4p)(uint32_t)(uintptr_t)p;           // LDS offset in low 32 bits
}

// ---------------------------------------------------------------------------
// One-pass conversion kernels (8 elems / thread, fully coalesced)
// ---------------------------------------------------------------------------
__global__ __launch_bounds__(256)
void convert_w_kernel(const int* __restrict__ q, half_t* __restrict__ w)
{
    size_t i = (size_t)blockIdx.x * blockDim.x + threadIdx.x;
    const int4* q4 = reinterpret_cast<const int4*>(q);
    int4 a = q4[2 * i];
    int4 b = q4[2 * i + 1];
    v8h o;
    o[0] = (_Float16)a.x; o[1] = (_Float16)a.y;
    o[2] = (_Float16)a.z; o[3] = (_Float16)a.w;
    o[4] = (_Float16)b.x; o[5] = (_Float16)b.y;
    o[6] = (_Float16)b.z; o[7] = (_Float16)b.w;
    *reinterpret_cast<v8h*>(w + 8 * i) = o;
}

__global__ __launch_bounds__(256)
void convert_x_kernel(const float* __restrict__ x, half_t* __restrict__ xh)
{
    size_t i = (size_t)blockIdx.x * blockDim.x + threadIdx.x;
    const float4* x4 = reinterpret_cast<const float4*>(x);
    float4 a = x4[2 * i];
    float4 b = x4[2 * i + 1];
    v8h o;
    o[0] = (_Float16)a.x; o[1] = (_Float16)a.y;
    o[2] = (_Float16)a.z; o[3] = (_Float16)a.w;
    o[4] = (_Float16)b.x; o[5] = (_Float16)b.y;
    o[6] = (_Float16)b.z; o[7] = (_Float16)b.w;
    *reinterpret_cast<v8h*>(xh + 8 * i) = o;
}

// ---------------------------------------------------------------------------
// WMMA GEMM: 256 threads = 8 waves (4x2); block tile 256x128; wave tile 64x64
// Double-buffered LDS; async global->LDS when available.
// ---------------------------------------------------------------------------
__global__ __launch_bounds__(256)
void int8linear_wmma_gemm(const half_t* __restrict__ Xh,   // [M,K] fp16
                          const half_t* __restrict__ Wh,   // [N,K] fp16
                          const float*  __restrict__ scale_p,
                          const float*  __restrict__ bias,
                          float*        __restrict__ out)  // [M,N] fp32
{
    __shared__ half_t sX[2][BM][LDSTR];   // 40 KB
    __shared__ half_t sW[2][BN][LDSTR];   // 20 KB

    const int tid  = threadIdx.x;
    const int lane = tid & 31;
    const int wid  = tid >> 5;
    const int h    = lane >> 4;      // half-wave (K-interleave group)
    const int ln   = lane & 15;      // row/col within a 16-wide tile

    const int bm = blockIdx.y * BM;
    const int bn = blockIdx.x * BN;

    const int wm = (wid >> 1) * 64;  // wave M offset (4 rows of waves)
    const int wn = (wid & 1) * 64;   // wave N offset (2 cols of waves)

    // staging assignment:
    //   X tile 256x32: thread t copies row t, 4 x b128
    //   W tile 128x32: thread t copies row t/2, 2 x b128 (half-row (t&1))
    const int wrow = tid >> 1;
    const int wcol = (tid & 1) * 16;                     // 0 or 16 halves
    const half_t* gx = Xh + (size_t)(bm + tid)  * K_DIM;
    const half_t* gw = Wh + (size_t)(bn + wrow) * K_DIM + wcol;

    auto stage = [&](int kk, int buf) {
#if USE_ASYNC_LDS
        #pragma unroll
        for (int s = 0; s < 4; ++s)
            __builtin_amdgcn_global_load_async_to_lds_b128(
                as_g4(gx + kk + s * 8), as_l4(&sX[buf][tid][s * 8]),
                /*offset=*/0, /*cpol=*/0);
        #pragma unroll
        for (int s = 0; s < 2; ++s)
            __builtin_amdgcn_global_load_async_to_lds_b128(
                as_g4(gw + kk + s * 8), as_l4(&sW[buf][wrow][wcol + s * 8]),
                /*offset=*/0, /*cpol=*/0);
#else
        v8h xr[4], wr[2];
        #pragma unroll
        for (int s = 0; s < 4; ++s)
            xr[s] = *reinterpret_cast<const v8h*>(gx + kk + s * 8);
        #pragma unroll
        for (int s = 0; s < 2; ++s)
            wr[s] = *reinterpret_cast<const v8h*>(gw + kk + s * 8);
        #pragma unroll
        for (int s = 0; s < 4; ++s)
            *reinterpret_cast<v8h*>(&sX[buf][tid][s * 8]) = xr[s];
        #pragma unroll
        for (int s = 0; s < 2; ++s)
            *reinterpret_cast<v8h*>(&sW[buf][wrow][wcol + s * 8]) = wr[s];
#endif
    };

    v8f acc[4][4];
    #pragma unroll
    for (int i = 0; i < 4; ++i)
        #pragma unroll
        for (int j = 0; j < 4; ++j)
            acc[i][j] = (v8f)(0.0f);

    auto compute = [&](int buf) {
        // A: lane<16 -> K {0..7,16..23}; lane>=16 -> K {8..15,24..31}
        v16h A[4];
        #pragma unroll
        for (int mt = 0; mt < 4; ++mt) {
            const int r = wm + mt * 16 + ln;
            v8h lo = *reinterpret_cast<const v8h*>(&sX[buf][r][h * 8]);
            v8h hi = *reinterpret_cast<const v8h*>(&sX[buf][r][16 + h * 8]);
            A[mt] = __builtin_shufflevector(lo, hi, 0,1,2,3,4,5,6,7,
                                                    8,9,10,11,12,13,14,15);
        }
        // B: lane holds column n=ln; lane<16 -> K {0..7,16..23}; lane>=16 -> K {8..15,24..31}
        v16h B[4];
        #pragma unroll
        for (int nt = 0; nt < 4; ++nt) {
            const int c = wn + nt * 16 + ln;
            v8h lo = *reinterpret_cast<const v8h*>(&sW[buf][c][h * 8]);
            v8h hi = *reinterpret_cast<const v8h*>(&sW[buf][c][16 + h * 8]);
            B[nt] = __builtin_shufflevector(lo, hi, 0,1,2,3,4,5,6,7,
                                                    8,9,10,11,12,13,14,15);
        }
        #pragma unroll
        for (int mt = 0; mt < 4; ++mt)
            #pragma unroll
            for (int nt = 0; nt < 4; ++nt)
                acc[mt][nt] = __builtin_amdgcn_wmma_f32_16x16x32_f16(
                    false, A[mt], false, B[nt],
                    (short)0, acc[mt][nt], false, false);
    };

    // prologue: fill buffer 0
    stage(0, 0);
#if USE_ASYNC_LDS
    wait_asynccnt0();
#endif
    __syncthreads();

    int buf = 0;
    for (int kk = 0; kk < K_DIM; kk += BK) {
        if (kk + BK < K_DIM)
            stage(kk + BK, buf ^ 1);   // copy overlaps compute (disjoint buffer)
        compute(buf);
#if USE_ASYNC_LDS
        wait_asynccnt0();              // this wave's async copies landed
#endif
        __syncthreads();               // everyone's copies visible; reads done
        buf ^= 1;
    }

    // epilogue: out = scale*acc + bias  (dequant folded here)
    const float scale = *scale_p;
    #pragma unroll
    for (int nt = 0; nt < 4; ++nt) {
        const int n  = bn + wn + nt * 16 + ln;
        const float bv = bias[n];
        #pragma unroll
        for (int mt = 0; mt < 4; ++mt) {
            const int mbase = bm + wm + mt * 16 + h * 8;  // VGPR v -> M = v + 8h
            #pragma unroll
            for (int v = 0; v < 8; ++v)
                out[(size_t)(mbase + v) * N_DIM + n] = scale * acc[mt][nt][v] + bv;
        }
    }
}

// ---------------------------------------------------------------------------
// launch
// ---------------------------------------------------------------------------
extern "C" void kernel_launch(void* const* d_in, const int* in_sizes, int n_in,
                              void* d_out, int out_size, void* d_ws, size_t ws_size,
                              hipStream_t stream)
{
    const float* x     = (const float*)d_in[0];
    const int*   qw    = (const int*)d_in[1];
    const float* scale = (const float*)d_in[2];
    const float* bias  = (const float*)d_in[3];
    float*       out   = (float*)d_out;

    // workspace: [0, 32MB) = W fp16, [32MB, 160MB) = X fp16
    half_t* w16 = (half_t*)d_ws;
    half_t* x16 = (half_t*)((char*)d_ws + (size_t)N_DIM * K_DIM * sizeof(half_t));

    {
        const size_t elems = (size_t)N_DIM * K_DIM;
        dim3 grid((unsigned)(elems / 8 / 256));
        convert_w_kernel<<<grid, 256, 0, stream>>>(qw, w16);
    }
    {
        const size_t elems = (size_t)M_DIM * K_DIM;
        dim3 grid((unsigned)(elems / 8 / 256));
        convert_x_kernel<<<grid, 256, 0, stream>>>(x, x16);
    }
    {
        dim3 grid(N_DIM / BN, M_DIM / BM);   // 32 x 64
        int8linear_wmma_gemm<<<grid, 256, 0, stream>>>(x16, w16, scale, bias, out);
    }
}
